// DETRHead_16724602650670
// MI455X (gfx1250) — matrix operand order repaired and checked by reading, not compiled, not run on hardware
//
#include <hip/hip_runtime.h>
#include <hip/hip_bf16.h>
#include <math.h>

// Problem constants
#define B_    32
#define L_    40
#define LP_   48            // text tokens padded to 3 x 16 N-tiles
#define NP_   2304
#define Q_    100
#define P_    2404          // NP_ + Q_
#define CIN_  768
#define COUT_ 256
#define MT_   151           // ceil(P_/16) M-tiles for the cos GEMM

// LDS staging for k_cos: 3 K-stages of 256 bf16 elems; padded row stride.
#define KSTAGE_   256
#define NSTAGE_   3
#define LDSROWB_  528        // 256*2 + 16 pad bytes -> 4-bank row skew, conflict-free
#define LDSBYTES_ (LP_ * LDSROWB_)   // 25344 per buffer; double-buffered

typedef __attribute__((ext_vector_type(16))) __bf16 v16bf;
typedef __attribute__((ext_vector_type(8)))  __bf16 v8bf;
typedef __attribute__((ext_vector_type(8)))  float  v8f;

#define NEG_INF (-__builtin_inff())

#if defined(__HIP_DEVICE_COMPILE__) && __has_builtin(__builtin_amdgcn_tensor_load_to_lds) && __has_builtin(__builtin_amdgcn_s_wait_tensorcnt)
#define HAVE_TDM 1
#else
#define HAVE_TDM 0
#endif

__device__ __forceinline__ __bf16 f2bf(float f) {
  // round-to-nearest-even f32 -> bf16 (prep kernels only)
  unsigned u = __builtin_bit_cast(unsigned, f);
  u += 0x7fffu + ((u >> 16) & 1u);
  unsigned short s = (unsigned short)(u >> 16);
  return __builtin_bit_cast(__bf16, s);
}

__device__ __forceinline__ float wave_sum(float v) {
#pragma unroll
  for (int m = 16; m >= 1; m >>= 1) v += __shfl_xor(v, m, 32);
  return v;
}

__device__ __forceinline__ v16bf cat8(v8bf lo, v8bf hi) {
  return __builtin_shufflevector(lo, hi, 0, 1, 2, 3, 4, 5, 6, 7,
                                 8, 9, 10, 11, 12, 13, 14, 15);
}

#if HAVE_TDM
typedef unsigned int u32x4 __attribute__((ext_vector_type(4)));
typedef int          i32x8 __attribute__((ext_vector_type(8)));
typedef int          i32x4 __attribute__((ext_vector_type(4)));

// TDM: load one 2-D tile (256 x 48 bf16 elems, tensor row stride 768 elems)
// into LDS with 16-byte padding after every 512-byte row (pad_interval=128
// DWORDs -> code 6, pad_amount=4 DWORDs -> code 3).  D# per CDNA5 ISA 8.3-8.5.
__device__ __forceinline__ void tdm_load_text_stage(const __bf16* gsrc, unsigned lds_off) {
  unsigned long long ga = (unsigned long long)(uintptr_t)gsrc;
  u32x4 g0;
  g0[0] = 1u;                                                 // count=1, user load
  g0[1] = lds_off;                                            // lds_addr (bytes)
  g0[2] = (unsigned)(ga & 0xffffffffu);                       // global_addr lo
  g0[3] = (unsigned)((ga >> 32) & 0x01ffffffu) | 0x80000000u; // addr hi | type=2
  i32x8 g1;
  g1[0] = (int)((1u << 16) | (1u << 20) | (6u << 22) | (3u << 25)); // dsize=2B, pad on
  g1[1] = (int)(768u << 16);   // tensor_dim0 = 768 (bits 79:48 low half)
  g1[2] = (int)(48u << 16);    // tensor_dim1 = 48
  g1[3] = (int)(256u << 16);   // tile_dim0 = 256
  g1[4] = 48;                  // tile_dim1 = 48
  g1[5] = 768;                 // tensor_dim0_stride = 768 elems
  g1[6] = 0;
  g1[7] = 0;
  i32x4 z = {0, 0, 0, 0};
#if __clang_major__ >= 23
  i32x8 z8 = {0, 0, 0, 0, 0, 0, 0, 0};
  __builtin_amdgcn_tensor_load_to_lds(g0, g1, z, z, z8, 0);
#else
  __builtin_amdgcn_tensor_load_to_lds(g0, g1, z, z, 0);
#endif
}
#endif

// ---------------------------------------------------------------------------
// Prep 1: L2-normalize text rows -> bf16, zero-pad L 40->48. One wave per row.
__global__ void k_text_prep(const float* __restrict__ text, __bf16* __restrict__ tb) {
  int wave = threadIdx.x >> 5, lane = threadIdx.x & 31;
  int row = blockIdx.x * 8 + wave;
  if (row >= B_ * LP_) return;
  int b = row / LP_, n = row % LP_;
  __bf16* orow = tb + (size_t)row * CIN_;
  if (n >= L_) {
#pragma unroll
    for (int j = 0; j < CIN_ / 32; ++j) orow[lane + j * 32] = f2bf(0.f);
    return;
  }
  const float* irow = text + ((size_t)b * L_ + n) * CIN_;
  float vals[CIN_ / 32];
  float ss = 0.f;
#pragma unroll
  for (int j = 0; j < CIN_ / 32; ++j) {
    float v = irow[lane + j * 32];
    vals[j] = v;
    ss += v * v;
  }
  ss = wave_sum(ss);
  float inv = 1.f / fmaxf(sqrtf(ss), 1e-8f);
#pragma unroll
  for (int j = 0; j < CIN_ / 32; ++j) orow[lane + j * 32] = f2bf(vals[j] * inv);
}

// ---------------------------------------------------------------------------
// Prep 2: fused L2-norm + bf16 conversion for img rows and the 100 dummy rows.
__global__ void k_imgall_prep(const float* __restrict__ img, const float* __restrict__ dummy,
                              __bf16* __restrict__ img_bf, __bf16* __restrict__ dummy_bf) {
  int wave = threadIdx.x >> 5, lane = threadIdx.x & 31;
  int row = blockIdx.x * 8 + wave;
  const int NROWS = B_ * NP_ + Q_;
  if (row >= NROWS) return;
  const float* rp;
  __bf16* op;
  if (row < B_ * NP_) { rp = img + (size_t)row * CIN_;                op = img_bf + (size_t)row * CIN_; }
  else                { rp = dummy + (size_t)(row - B_ * NP_) * CIN_; op = dummy_bf + (size_t)(row - B_ * NP_) * CIN_; }
  float vals[CIN_ / 32];
  float ss = 0.f;
#pragma unroll
  for (int j = 0; j < CIN_ / 32; ++j) {
    float v = rp[lane + j * 32];
    vals[j] = v;
    ss += v * v;
  }
  ss = wave_sum(ss);
  float inv = 1.f / fmaxf(sqrtf(ss), 1e-8f);
#pragma unroll
  for (int j = 0; j < CIN_ / 32; ++j) op[lane + j * 32] = f2bf(vals[j] * inv);
}

// ---------------------------------------------------------------------------
// Prep 3: Wq f32 -> bf16.
__global__ void k_wq_prep(const float* __restrict__ wq, __bf16* __restrict__ wqb) {
  int i = blockIdx.x * blockDim.x + threadIdx.x;
  if (i < COUT_ * CIN_) wqb[i] = f2bf(wq[i]);
}

// ---------------------------------------------------------------------------
// Cos-similarity GEMM + scale/mask + row-max.  One wave = 16(M) x 48(N) tile.
// B operand (text block for this batch) staged to LDS via a DOUBLE-BUFFERED
// async TDM pipeline: issue stage s+1, wait tensorcnt<=1 (in-order per wave),
// barrier, compute stage s from the other buffer.
__global__ __launch_bounds__(256) void k_cos(
    const __bf16* __restrict__ img_bf, const __bf16* __restrict__ dummy_bf,
    const __bf16* __restrict__ tb,
    const float* __restrict__ scores, const int* __restrict__ mask,
    float* __restrict__ sim_out, float* __restrict__ scl_out,
    float* __restrict__ maxpp) {
  __shared__ __align__(16) unsigned char smem[2 * LDSBYTES_];

  int wave = threadIdx.x >> 5, lane = threadIdx.x & 31;
  int mtile_raw = blockIdx.x * 8 + wave;
  int mtile = (mtile_raw < MT_) ? mtile_raw : MT_ - 1;  // clamp: keep barriers uniform
  int b = blockIdx.y;
  int lane16 = lane & 15, half = lane >> 4;

  // A operand: this lane owns matrix row m = lane16 (normalized bf16).
  int grow = mtile * 16 + lane16;
  int growc = (grow < P_) ? grow : 0;
  const __bf16* arow = (growc < NP_)
      ? img_bf + ((size_t)b * NP_ + growc) * CIN_
      : dummy_bf + (size_t)(growc - NP_) * CIN_;
  const __bf16* tbat = tb + (size_t)b * LP_ * CIN_;

#if HAVE_TDM
  unsigned ldsbase = (unsigned)(uintptr_t)&smem[0];
  if (threadIdx.x < 32) {
    tdm_load_text_stage(tbat, ldsbase);          // prologue: stage 0 -> buf 0
  }
#endif

  v8f acc0 = {}, acc1 = {}, acc2 = {};
  for (int s = 0; s < NSTAGE_; ++s) {
    const unsigned char* buf = smem + (unsigned)(s & 1) * LDSBYTES_;
#if HAVE_TDM
    if (threadIdx.x < 32) {
      if (s + 1 < NSTAGE_) {
        tdm_load_text_stage(tbat + (s + 1) * KSTAGE_,
                            ldsbase + (unsigned)((s + 1) & 1) * LDSBYTES_);
        // tensor ops from one wave complete in order: <=1 pending => stage s done
        __builtin_amdgcn_s_wait_tensorcnt(1);
      } else {
        __builtin_amdgcn_s_wait_tensorcnt(0);
      }
    }
#else
    for (int c = (int)threadIdx.x; c < LP_ * 32; c += 256) {
      int rrow = c >> 5, c16 = c & 31;
      *(uint4*)((unsigned char*)buf + rrow * LDSROWB_ + c16 * 16) =
          *(const uint4*)((const unsigned char*)(tbat + (size_t)rrow * CIN_ + s * KSTAGE_) + c16 * 16);
    }
#endif
    __syncthreads();  // stage s visible to all waves

    for (int kk = 0; kk < KSTAGE_; kk += 32) {
      int k0 = s * KSTAGE_ + kk;
      // A frag: lanes 0-15 K=k0+{0..15}; lanes 16-31 K=k0+{16..31}
      int kA = k0 + half * 16;
      v8bf alo = *(const v8bf*)(arow + kA);
      v8bf ahi = *(const v8bf*)(arow + kA + 8);
      v16bf a = cat8(alo, ahi);

      // B frags from LDS: col n = t*16+lane16, 16 K-contiguous bf16
      int koff = (kk + half * 16) * 2;
      const unsigned char* p0 = buf + (size_t)(0 * 16 + lane16) * LDSROWB_ + koff;
      const unsigned char* p1 = buf + (size_t)(1 * 16 + lane16) * LDSROWB_ + koff;
      const unsigned char* p2 = buf + (size_t)(2 * 16 + lane16) * LDSROWB_ + koff;
      v16bf b0 = cat8(*(const v8bf*)p0, *(const v8bf*)(p0 + 16));
      v16bf b1 = cat8(*(const v8bf*)p1, *(const v8bf*)(p1 + 16));
      v16bf b2 = cat8(*(const v8bf*)p2, *(const v8bf*)(p2 + 16));

      acc0 = __builtin_amdgcn_wmma_f32_16x16x32_bf16(false, a, false, b0, (short)0, acc0, false, false);
      acc1 = __builtin_amdgcn_wmma_f32_16x16x32_bf16(false, a, false, b1, (short)0, acc1, false, false);
      acc2 = __builtin_amdgcn_wmma_f32_16x16x32_bf16(false, a, false, b2, (short)0, acc2, false, false);
    }
    __syncthreads();  // all reads of buf done before it is overwritten
  }

  float log_term = logf(1.22f - scores[b]);
  float sc_real = -0.59f * log_term + 0.12f;
  float sc_dum  =  0.59f * log_term + 0.88f;
  const int* mrow = mask + b * L_;

  // C layout: VGPR r, lanes 0-15 -> M=r, lanes 16-31 -> M=r+8; N = lane16.
#pragma unroll
  for (int r = 0; r < 8; ++r) {
    int row = mtile * 16 + r + half * 8;
    bool rvalid = row < P_;
    float rowmax = NEG_INF;
#pragma unroll
    for (int t = 0; t < 3; ++t) {
      float cv = (t == 0) ? acc0[r] : (t == 1) ? acc1[r] : acc2[r];
      int n = t * 16 + lane16;
      if (rvalid && n < L_) {
        float sim = (cv + 1.f) * 0.5f;
        size_t o = ((size_t)b * P_ + row) * L_ + n;
        sim_out[o] = sim;
        float scl = (row < NP_) ? sc_real : sc_dum;
        float sc = mrow[n] ? NEG_INF : scl * sim;
        scl_out[o] = sc;
        rowmax = fmaxf(rowmax, sc);
      }
    }
#pragma unroll
    for (int msk = 8; msk >= 1; msk >>= 1)
      rowmax = fmaxf(rowmax, __shfl_xor(rowmax, msk, 32));
    if (lane16 == 0 && rvalid) maxpp[(size_t)b * P_ + row] = rowmax;
  }
}

// ---------------------------------------------------------------------------
// Top-100 per batch via iterative argmax (tie -> lowest index, jax-compatible).
__global__ void k_topk(float* __restrict__ vals, int* __restrict__ idx_out) {
  int b = blockIdx.x, tid = threadIdx.x;
  float* v = vals + (size_t)b * P_;
  __shared__ float sv[256];
  __shared__ int si[256];
  for (int q = 0; q < Q_; ++q) {
    float bv = NEG_INF;
    int bi = P_;
    for (int i = tid; i < P_; i += 256) {
      float x = v[i];
      if (x > bv || (x == bv && i < bi)) { bv = x; bi = i; }
    }
    sv[tid] = bv; si[tid] = bi;
    __syncthreads();
    for (int s = 128; s > 0; s >>= 1) {
      if (tid < s) {
        if (sv[tid + s] > sv[tid] || (sv[tid + s] == sv[tid] && si[tid + s] < si[tid])) {
          sv[tid] = sv[tid + s]; si[tid] = si[tid + s];
        }
      }
      __syncthreads();
    }
    if (tid == 0) { idx_out[b * Q_ + q] = si[0]; v[si[0]] = NEG_INF; }
    __syncthreads();
  }
}

// ---------------------------------------------------------------------------
// Gather the 3200 selected rows (raw, un-normalized) and convert once to bf16.
__global__ void k_gather(const float* __restrict__ img, const float* __restrict__ dummy,
                         const int* __restrict__ idx, __bf16* __restrict__ sel_bf) {
  int wave = threadIdx.x >> 5, lane = threadIdx.x & 31;
  int gm = blockIdx.x * 8 + wave;
  if (gm >= B_ * Q_) return;
  int b = gm / Q_, q = gm % Q_;
  int sel = idx[b * Q_ + q];
  sel = (sel < 0) ? 0 : (sel >= P_ ? P_ - 1 : sel);
  const float* src = (sel < NP_) ? img + ((size_t)b * NP_ + sel) * CIN_
                                 : dummy + (size_t)(sel - NP_) * CIN_;
  __bf16* dst = sel_bf + (size_t)gm * CIN_;
#pragma unroll
  for (int j = 0; j < CIN_ / 32; ++j) dst[lane + j * 32] = f2bf(src[lane + j * 32]);
}

// ---------------------------------------------------------------------------
// query = selected @ Wq^T + bq.  M=3200, N=256, K=768.
// One wave: 16-row M-tile x 4 consecutive N-tiles (A frag reused 4x).
__global__ __launch_bounds__(256) void k_query(
    const __bf16* __restrict__ sel_bf, const __bf16* __restrict__ wqb,
    const float* __restrict__ bq, float* __restrict__ out) {
  int wave = threadIdx.x >> 5, lane = threadIdx.x & 31;
  int lane16 = lane & 15, half = lane >> 4;
  int mtile = blockIdx.x * 2 + (wave >> 2);   // 0..199
  int ntb = (wave & 3) * 4;                   // first of 4 n-tiles

  const __bf16* arow = sel_bf + (size_t)(mtile * 16 + lane16) * CIN_;

  v8f acc0 = {}, acc1 = {}, acc2 = {}, acc3 = {};
  for (int k0 = 0; k0 < CIN_; k0 += 32) {
    int kA = k0 + half * 16;
    v16bf a = cat8(*(const v8bf*)(arow + kA), *(const v8bf*)(arow + kA + 8));

    const __bf16* wbase = wqb + k0 + half * 16;
    v16bf b0 = *(const v16bf*)(wbase + (size_t)((ntb + 0) * 16 + lane16) * CIN_);
    v16bf b1 = *(const v16bf*)(wbase + (size_t)((ntb + 1) * 16 + lane16) * CIN_);
    v16bf b2 = *(const v16bf*)(wbase + (size_t)((ntb + 2) * 16 + lane16) * CIN_);
    v16bf b3 = *(const v16bf*)(wbase + (size_t)((ntb + 3) * 16 + lane16) * CIN_);

    acc0 = __builtin_amdgcn_wmma_f32_16x16x32_bf16(false, a, false, b0, (short)0, acc0, false, false);
    acc1 = __builtin_amdgcn_wmma_f32_16x16x32_bf16(false, a, false, b1, (short)0, acc1, false, false);
    acc2 = __builtin_amdgcn_wmma_f32_16x16x32_bf16(false, a, false, b2, (short)0, acc2, false, false);
    acc3 = __builtin_amdgcn_wmma_f32_16x16x32_bf16(false, a, false, b3, (short)0, acc3, false, false);
  }

#pragma unroll
  for (int t = 0; t < 4; ++t) {
    int n = (ntb + t) * 16 + lane16;
    float bqv = bq[n];
#pragma unroll
    for (int r = 0; r < 8; ++r) {
      float cv = (t == 0) ? acc0[r] : (t == 1) ? acc1[r] : (t == 2) ? acc2[r] : acc3[r];
      int row = mtile * 16 + r + half * 8;
      out[(size_t)row * COUT_ + n] = cv + bqv;
    }
  }
}

// ---------------------------------------------------------------------------
extern "C" void kernel_launch(void* const* d_in, const int* in_sizes, int n_in,
                              void* d_out, int out_size, void* d_ws, size_t ws_size,
                              hipStream_t stream) {
  (void)in_sizes; (void)n_in; (void)out_size; (void)ws_size;

  const float* text   = (const float*)d_in[0];   // [32,40,768]
  const float* img    = (const float*)d_in[1];   // [32,2304,768]
  const float* scores = (const float*)d_in[2];   // [32,1]
  const int*   mask   = (const int*)d_in[3];     // [32,40]
  const float* dummy  = (const float*)d_in[4];   // [1,100,768]
  const float* wq     = (const float*)d_in[5];   // [256,768]
  const float* bq     = (const float*)d_in[6];   // [256]

  // Workspace layout (256-B aligned), ~121 MB
  char* ws = (char*)d_ws;
  __bf16* text_bf  = (__bf16*)(ws + 0);          // 1536*768
  __bf16* wq_bf    = (__bf16*)(ws + 2359296);    // 256*768
  __bf16* dummy_bf = (__bf16*)(ws + 2752512);    // 100*768
  float*  maxpp    = (float*)(ws + 2906112);     // 32*2404
  __bf16* sel_bf   = (__bf16*)(ws + 3213824);    // 3200*768
  __bf16* img_bf   = (__bf16*)(ws + 8129024);    // 32*2304*768

  // Output: query | topk_idx | sim_norm | scaled (flat concat, return order)
  float* out     = (float*)d_out;
  float* query   = out;                          // 819200
  int*   idx_out = (int*)(out + 819200);         // 3200 (int32 bits)
  float* sim_out = out + 822400;                 // 3077120
  float* scl_out = out + 3899520;                // 3077120

  k_text_prep<<<dim3(192), dim3(256), 0, stream>>>(text, text_bf);
  k_imgall_prep<<<dim3(9229), dim3(256), 0, stream>>>(img, dummy, img_bf, dummy_bf);
  k_wq_prep<<<dim3(768), dim3(256), 0, stream>>>(wq, wq_bf);
  k_cos<<<dim3(19, 32), dim3(256), 0, stream>>>(img_bf, dummy_bf, text_bf,
                                                scores, mask, sim_out, scl_out, maxpp);
  k_topk<<<dim3(32), dim3(256), 0, stream>>>(maxpp, idx_out);
  k_gather<<<dim3(400), dim3(256), 0, stream>>>(img, dummy, idx_out, sel_bf);
  k_query<<<dim3(100), dim3(256), 0, stream>>>(sel_bf, wq_bf, bq, query);
}
